// EventAwareVisual_52441550684693
// MI455X (gfx1250) — matrix operand order repaired and checked by reading, not compiled, not run on hardware
//
#include <hip/hip_runtime.h>
#include <hip/hip_bf16.h>
#include <math.h>

typedef __attribute__((ext_vector_type(16))) __bf16 bf16x16;
typedef __attribute__((ext_vector_type(8)))  __bf16 bf16x8;
typedef __attribute__((ext_vector_type(8)))  float  f32x8;

#define DMODEL 128
#define NB     32
#define SV     512
#define NH     4
#define DHD    32
#define ATT_SCALE 0.17677669529663687f  // 1/sqrt(32)

// ---------------------------------------------------------------------------
// WMMA helpers (CDNA5 bf16 16x16x32, fp32 accumulate)
// ---------------------------------------------------------------------------
static __device__ __forceinline__ f32x8 wmma_bf16(bf16x16 a, bf16x16 b, f32x8 c) {
  return __builtin_amdgcn_wmma_f32_16x16x32_bf16(false, a, false, b, (short)0, c,
                                                 false, false);
}

// A fragment (16x32 bf16): lane<16 holds K {0..15}, lane>=16 holds {16..31}.
// base points at row + kc + half; loads 16B at +0 and 16B at +8 elements.
static __device__ __forceinline__ bf16x16 load_afrag(const __bf16* base) {
  union { bf16x16 v16; bf16x8 v8[2]; } u;
  u.v8[0] = *(const bf16x8*)(base);
  u.v8[1] = *(const bf16x8*)(base + 8);
  return u.v16;
}

// ---------------------------------------------------------------------------
// Generic N=128 bf16 GEMM:  C[M,128] = act( A[M,K] @ Bt[128,K]^T + bias + constRow )
// Block = 128 threads = 4 waves; wave computes one 16-row strip x 128 cols
// (8 accumulator tiles, A fragment reused across all 8 N tiles).
// ---------------------------------------------------------------------------
__global__ __launch_bounds__(128) void gemm_n128(
    const __bf16* __restrict__ A, int lda, int K,
    const __bf16* __restrict__ Bt,            // [128][K] = W^T, bf16
    __bf16* __restrict__ C,
    const float* __restrict__ bias,           // [128] or null
    const float* __restrict__ constRow,       // [NB][128] or null (indexed by m>>9)
    int do_relu, int vt_store)
{
  const int lane = threadIdx.x & 31;
  const int wave = threadIdx.x >> 5;
  const int m0   = blockIdx.x * 64 + wave * 16;
  const int l15  = lane & 15;
  const int ah   = (lane < 16) ? 0 : 16;
  const int bh   = (lane < 16) ? 0 : 16;

  f32x8 acc[8] = {};
  const __bf16* arow = A + (size_t)(m0 + l15) * lda;

  for (int kc = 0; kc < K; kc += 32) {
    bf16x16 a = load_afrag(arow + kc + ah);
#pragma unroll
    for (int t = 0; t < 8; ++t) {
      bf16x16 b = *(const bf16x16*)(Bt + (size_t)(t * 16 + l15) * K + kc + bh);
      acc[t] = wmma_bf16(a, b, acc[t]);
    }
  }

  const int mb = m0 + ((lane >= 16) ? 8 : 0);
#pragma unroll
  for (int t = 0; t < 8; ++t) {
    const int n = t * 16 + l15;
#pragma unroll
    for (int i = 0; i < 8; ++i) {
      const int m = mb + i;
      float v = acc[t][i];
      if (bias)     v += bias[n];
      if (constRow) v += constRow[(m >> 9) * DMODEL + n];
      if (do_relu)  v = fmaxf(v, 0.0f);
      if (vt_store) {
        // store transposed per-head: vT[b][h][dh][seq]
        const int bb = m >> 9, seq = m & 511, h = n >> 5, dh = n & 31;
        C[(((size_t)(bb * NH + h) * DHD + dh) << 9) + seq] = (__bf16)v;
      } else {
        C[(size_t)m * DMODEL + n] = (__bf16)v;
      }
    }
  }
}

// ---------------------------------------------------------------------------
// Fused self-attention: one wave per (b, h, 16-query strip).
// S = (q k^T)/sqrt(32) -> softmax -> O = P v   (P kept in LDS as bf16)
// ---------------------------------------------------------------------------
__global__ __launch_bounds__(32) void flash_attn(
    const __bf16* __restrict__ q, const __bf16* __restrict__ k,
    const __bf16* __restrict__ vT, __bf16* __restrict__ O)
{
  __shared__ __align__(16) float  S[16 * 512];
  __shared__ __align__(32) __bf16 P[16 * 512];

  const int lane  = threadIdx.x;
  const int idx   = blockIdx.x;
  const int strip = idx & 31;
  const int h     = (idx >> 5) & 3;
  const int b     = idx >> 7;
  const int l15   = lane & 15;
  const int ah    = (lane < 16) ? 0 : 16;
  const int bh    = (lane < 16) ? 0 : 16;
  const int mb    = (lane >= 16) ? 8 : 0;

  const __bf16* qrow = q + (size_t)(b * SV + strip * 16 + l15) * DMODEL + h * DHD;
  bf16x16 aq = load_afrag(qrow + ah);

  for (int kt = 0; kt < 32; ++kt) {
    bf16x16 bk = *(const bf16x16*)(k + (size_t)(b * SV + kt * 16 + l15) * DMODEL
                                     + h * DHD + bh);
    f32x8 z = {};
    f32x8 s = wmma_bf16(aq, bk, z);
#pragma unroll
    for (int i = 0; i < 8; ++i)
      S[(mb + i) * 512 + kt * 16 + l15] = s[i] * ATT_SCALE;
  }
  __syncthreads();

  // softmax: lanes l and l+16 share row l15, each covers 256 keys
  const int r  = l15;
  const int c0 = (lane >> 4) * 256;
  float mx = -1e30f;
  for (int c = 0; c < 256; ++c) mx = fmaxf(mx, S[r * 512 + c0 + c]);
  mx = fmaxf(mx, __shfl_xor(mx, 16, 32));
  float sum = 0.0f;
  for (int c = 0; c < 256; ++c) {
    float e = __expf(S[r * 512 + c0 + c] - mx);
    sum += e;
    P[r * 512 + c0 + c] = (__bf16)e;
  }
  sum += __shfl_xor(sum, 16, 32);
  const float inv = 1.0f / sum;
  for (int c = 0; c < 256; ++c) {
    const int e = r * 512 + c0 + c;
    P[e] = (__bf16)((float)P[e] * inv);
  }
  __syncthreads();

  // O strip = P @ v  (N = 32 -> two 16-col tiles, B from transposed v)
  f32x8 acc0 = {}, acc1 = {};
  const __bf16* prow = &P[r * 512];
  const __bf16* vbase = vT + (size_t)((b * NH + h) * DHD) * SV;
  for (int kc = 0; kc < 512; kc += 32) {
    bf16x16 ap = load_afrag(prow + kc + ah);
    bf16x16 b0 = *(const bf16x16*)(vbase + (size_t)(l15) * SV + kc + bh);
    bf16x16 b1 = *(const bf16x16*)(vbase + (size_t)(16 + l15) * SV + kc + bh);
    acc0 = wmma_bf16(ap, b0, acc0);
    acc1 = wmma_bf16(ap, b1, acc1);
  }
#pragma unroll
  for (int i = 0; i < 8; ++i) {
    const size_t row = (size_t)(b * SV + strip * 16 + mb + i) * DMODEL + h * DHD;
    O[row + l15]      = (__bf16)acc0[i];
    O[row + 16 + l15] = (__bf16)acc1[i];
  }
}

// ---------------------------------------------------------------------------
// Prep kernels
// ---------------------------------------------------------------------------
__global__ void cvt_bf16(const float* __restrict__ src, __bf16* __restrict__ dst, int n) {
  int i = blockIdx.x * 256 + threadIdx.x;
  if (i < n) dst[i] = (__bf16)src[i];
}

// dst[n*K + k] = (bf16) src[(k+rowOff)*N + n]   (transpose-convert weights)
__global__ void tconv_bf16(const float* __restrict__ src, __bf16* __restrict__ dst,
                           int K, int N, int rowOff) {
  int i = blockIdx.x * 256 + threadIdx.x;
  if (i < K * N) {
    int n = i / K, kk = i % K;
    dst[i] = (__bf16)src[(size_t)(kk + rowOff) * N + n];
  }
}

// Per-batch: h_e = relu(event@W_e+b_e); const_e = h_e@W_proj[0:128]+b_proj;
// cq = h_e@mha_Wq; h_v_mean = cq@mha_Wov  (exact collapse of transpose-attention)
__global__ __launch_bounds__(128) void event_prep(
    const float* __restrict__ event_fea, const float* __restrict__ W_e,
    const float* __restrict__ b_e, const float* __restrict__ W_proj,
    const float* __restrict__ b_proj, const float* __restrict__ mha_Wq,
    const float* __restrict__ mha_Wov,
    float* __restrict__ const_e, float* __restrict__ cq, float* __restrict__ hv_mean)
{
  __shared__ float he[128], cqs[128];
  const int b = blockIdx.x, n = threadIdx.x;
  float s = b_e[n];
  for (int kk = 0; kk < 512; ++kk) s += event_fea[b * 512 + kk] * W_e[kk * 128 + n];
  he[n] = fmaxf(s, 0.0f);
  __syncthreads();
  float c = b_proj[n], qv = 0.0f;
  for (int kk = 0; kk < 128; ++kk) {
    c  += he[kk] * W_proj[kk * 128 + n];
    qv += he[kk] * mha_Wq[kk * 128 + n];
  }
  const_e[b * 128 + n] = c;
  cqs[n] = qv;
  cq[b * 128 + n] = qv;
  __syncthreads();
  float hm = 0.0f;
  for (int kk = 0; kk < 128; ++kk) hm += cqs[kk] * mha_Wov[kk * 128 + n];
  hv_mean[b * 128 + n] = hm;
}

// Collapsed cross-attention: single query row per (b,h) ->
// ctx[b, h*32+dh] = sum_key softmax(cq.ck/sqrt32)[key] * cv[b,key,h*32+dh]
__global__ __launch_bounds__(256) void cross_ctx(
    const float* __restrict__ cq, const __bf16* __restrict__ ck,
    const __bf16* __restrict__ cv, float* __restrict__ ctx)
{
  __shared__ float sc[4 * 512];
  __shared__ float red[4 * 64];
  __shared__ float stat[8];
  const int b = blockIdx.x, t = threadIdx.x;
  for (int h = 0; h < 4; ++h) {
    const float* qr = cq + b * 128 + h * 32;
    for (int key = t; key < 512; key += 256) {
      const __bf16* kr = ck + (size_t)(b * 512 + key) * 128 + h * 32;
      float s = 0.0f;
      for (int d = 0; d < 32; ++d) s += qr[d] * (float)kr[d];
      sc[h * 512 + key] = s * ATT_SCALE;
    }
  }
  __syncthreads();
  const int h = t >> 6, sl = t & 63;
  float mx = -1e30f;
  for (int key = sl; key < 512; key += 64) mx = fmaxf(mx, sc[h * 512 + key]);
  red[h * 64 + sl] = mx;
  __syncthreads();
  if (sl == 0) {
    float m = red[h * 64];
    for (int j = 1; j < 64; ++j) m = fmaxf(m, red[h * 64 + j]);
    stat[h] = m;
  }
  __syncthreads();
  const float m = stat[h];
  float sum = 0.0f;
  for (int key = sl; key < 512; key += 64) {
    float e = __expf(sc[h * 512 + key] - m);
    sc[h * 512 + key] = e;
    sum += e;
  }
  red[h * 64 + sl] = sum;
  __syncthreads();
  if (sl == 0) {
    float s2 = 0.0f;
    for (int j = 0; j < 64; ++j) s2 += red[h * 64 + j];
    stat[4 + h] = s2;
  }
  __syncthreads();
  if (t < 128) {
    const int h2 = t >> 5, dh = t & 31;
    const float inv = 1.0f / stat[4 + h2];
    const __bf16* cvb = cv + (size_t)b * 512 * 128 + h2 * 32 + dh;
    float acc = 0.0f;
    for (int key = 0; key < 512; ++key)
      acc += sc[h2 * 512 + key] * (float)cvb[(size_t)key * 128];
    ctx[b * 128 + t] = acc * inv;
  }
}

static __device__ __forceinline__ float block_ln(float v, const float* g,
                                                 const float* bta, int n, float* red) {
  red[n] = v; __syncthreads();
  for (int s = 64; s > 0; s >>= 1) { if (n < s) red[n] += red[n + s]; __syncthreads(); }
  const float mu = red[0] / 128.0f; __syncthreads();
  const float d = v - mu;
  red[n] = d * d; __syncthreads();
  for (int s = 64; s > 0; s >>= 1) { if (n < s) red[n] += red[n + s]; __syncthreads(); }
  const float var = red[0] / 128.0f; __syncthreads();
  return d * rsqrtf(var + 1e-5f) * g[n] + bta[n];
}

// Tail: h_e_mean = ctx@mha_Wo; g_e_v, g_v (relu+LN); classifier -> out
__global__ __launch_bounds__(128) void tail_kernel(
    const float* __restrict__ ctx, const float* __restrict__ hv_mean,
    const float* __restrict__ mha_Wo,
    const float* __restrict__ W_ge, const float* __restrict__ b_ge,
    const float* __restrict__ W_gv, const float* __restrict__ b_gv,
    const float* __restrict__ ln_e_g, const float* __restrict__ ln_e_b,
    const float* __restrict__ ln_v_g, const float* __restrict__ ln_v_b,
    const float* __restrict__ W_c1, const float* __restrict__ b_c1,
    const float* __restrict__ W_c2, const float* __restrict__ b_c2,
    float* __restrict__ out)
{
  __shared__ float xe[128], buf[128], red[128];
  const int b = blockIdx.x, n = threadIdx.x;

  float a = 0.0f;
  for (int kk = 0; kk < 128; ++kk) a += ctx[b * 128 + kk] * mha_Wo[kk * 128 + n];
  xe[n] = a; __syncthreads();

  float ye = b_ge[n];
  for (int kk = 0; kk < 128; ++kk) ye += xe[kk] * W_ge[kk * 128 + n];
  ye = fmaxf(ye, 0.0f);
  const float gev = block_ln(ye, ln_e_g, ln_e_b, n, red);
  out[64 + 32 * 128 + b * 128 + n] = gev;   // g_e_v

  float yv = b_gv[n];
  for (int kk = 0; kk < 128; ++kk) yv += hv_mean[b * 128 + kk] * W_gv[kk * 128 + n];
  yv = fmaxf(yv, 0.0f);
  const float gvv = block_ln(yv, ln_v_g, ln_v_b, n, red);
  out[64 + b * 128 + n] = gvv;              // g_v
  buf[n] = gvv; __syncthreads();

  float c1 = b_c1[n];
  for (int kk = 0; kk < 128; ++kk) c1 += buf[kk] * W_c1[kk * 128 + n];
  c1 = fmaxf(c1, 0.0f);
  xe[n] = c1; __syncthreads();
  if (n < 2) {
    float o = b_c2[n];
    for (int kk = 0; kk < 128; ++kk) o += xe[kk] * W_c2[kk * 2 + n];
    out[b * 2 + n] = o;                     // output_local_v
  }
}

// ---------------------------------------------------------------------------
extern "C" void kernel_launch(void* const* d_in, const int* in_sizes, int n_in,
                              void* d_out, int out_size, void* d_ws, size_t ws_size,
                              hipStream_t stream) {
  (void)in_sizes; (void)n_in; (void)out_size; (void)ws_size;
  const float* visual  = (const float*)d_in[1];
  const float* event   = (const float*)d_in[2];
  const float* W_e     = (const float*)d_in[4];
  const float* b_e     = (const float*)d_in[5];
  const float* W_v     = (const float*)d_in[6];
  const float* b_v     = (const float*)d_in[7];
  const float* W_proj  = (const float*)d_in[8];
  const float* b_proj  = (const float*)d_in[9];
  const float* mma_Wq  = (const float*)d_in[10];
  const float* mma_Wk  = (const float*)d_in[11];
  const float* mma_Wv  = (const float*)d_in[12];
  const float* mma_Wo  = (const float*)d_in[13];
  const float* mha_Wq  = (const float*)d_in[14];
  const float* mha_Wk  = (const float*)d_in[15];
  const float* mha_Wv  = (const float*)d_in[16];
  const float* mha_Wo  = (const float*)d_in[17];
  const float* mha_Wov = (const float*)d_in[18];
  const float* W_ge    = (const float*)d_in[19];
  const float* b_ge    = (const float*)d_in[20];
  const float* W_gv    = (const float*)d_in[21];
  const float* b_gv    = (const float*)d_in[22];
  const float* ln_e_g  = (const float*)d_in[23];
  const float* ln_e_b  = (const float*)d_in[24];
  const float* ln_v_g  = (const float*)d_in[25];
  const float* ln_v_b  = (const float*)d_in[26];
  const float* W_c1    = (const float*)d_in[27];
  const float* b_c1    = (const float*)d_in[28];
  const float* W_c2    = (const float*)d_in[29];
  const float* b_c2    = (const float*)d_in[30];
  float* out = (float*)d_out;

  const size_t M  = (size_t)NB * SV;      // 16384
  const size_t AE = M * DMODEL;           // activation elements (2 bytes each)

  char* ws = (char*)d_ws;
  // 16MB region: visual_bf16 first, then reused as q | k | vT | O (4MB each)
  __bf16* vf  = (__bf16*)(ws);
  __bf16* qm  = (__bf16*)(ws);
  __bf16* km  = (__bf16*)(ws + AE * 2);
  __bf16* vT  = (__bf16*)(ws + AE * 4);
  __bf16* Om  = (__bf16*)(ws + AE * 6);
  __bf16* ckm = qm;                       // reuse after q is dead
  __bf16* cvm = km;                       // reuse after k is dead
  size_t off = AE * 8;
  __bf16* hv  = (__bf16*)(ws + off); off += AE * 2;
  __bf16* hf  = (__bf16*)(ws + off); off += AE * 2;   // also reused for hf2
  __bf16* WvT     = (__bf16*)(ws + off); off += 512 * 128 * 2;
  __bf16* WprojVT = (__bf16*)(ws + off); off += 128 * 128 * 2;
  __bf16* WqT     = (__bf16*)(ws + off); off += 128 * 128 * 2;
  __bf16* WkT     = (__bf16*)(ws + off); off += 128 * 128 * 2;
  __bf16* WvmT    = (__bf16*)(ws + off); off += 128 * 128 * 2;
  __bf16* WoT     = (__bf16*)(ws + off); off += 128 * 128 * 2;
  __bf16* WkcT    = (__bf16*)(ws + off); off += 128 * 128 * 2;
  __bf16* WvcT    = (__bf16*)(ws + off); off += 128 * 128 * 2;
  float* const_e  = (float*)(ws + off); off += NB * 128 * 4;
  float* cq       = (float*)(ws + off); off += NB * 128 * 4;
  float* hvmean   = (float*)(ws + off); off += NB * 128 * 4;
  float* ctx      = (float*)(ws + off); off += NB * 128 * 4;

  // --- prep: convert visual to bf16; transpose-convert weights ---
  {
    const int n = (int)(M * 512);
    cvt_bf16<<<(n + 255) / 256, 256, 0, stream>>>(visual, vf, n);
  }
  tconv_bf16<<<(512 * 128 + 255) / 256, 256, 0, stream>>>(W_v,     WvT,     512, 128, 0);
  tconv_bf16<<<(128 * 128 + 255) / 256, 256, 0, stream>>>(W_proj,  WprojVT, 128, 128, 128);
  tconv_bf16<<<(128 * 128 + 255) / 256, 256, 0, stream>>>(mma_Wq,  WqT,     128, 128, 0);
  tconv_bf16<<<(128 * 128 + 255) / 256, 256, 0, stream>>>(mma_Wk,  WkT,     128, 128, 0);
  tconv_bf16<<<(128 * 128 + 255) / 256, 256, 0, stream>>>(mma_Wv,  WvmT,    128, 128, 0);
  tconv_bf16<<<(128 * 128 + 255) / 256, 256, 0, stream>>>(mma_Wo,  WoT,     128, 128, 0);
  tconv_bf16<<<(128 * 128 + 255) / 256, 256, 0, stream>>>(mha_Wk,  WkcT,    128, 128, 0);
  tconv_bf16<<<(128 * 128 + 255) / 256, 256, 0, stream>>>(mha_Wv,  WvcT,    128, 128, 0);

  event_prep<<<NB, 128, 0, stream>>>(event, W_e, b_e, W_proj, b_proj,
                                     mha_Wq, mha_Wov, const_e, cq, hvmean);

  const int gGemm = (int)(M / 64);  // 256 blocks
  // h_v = relu(visual @ W_v + b_v)
  gemm_n128<<<gGemm, 128, 0, stream>>>(vf, 512, 512, WvT, hv, b_v, nullptr, 1, 0);
  // h_f = h_v @ W_proj[128:] + (h_e@W_proj[:128] + b_proj)
  gemm_n128<<<gGemm, 128, 0, stream>>>(hv, 128, 128, WprojVT, hf, nullptr, const_e, 0, 0);
  // q, k, v (v stored transposed per head)
  gemm_n128<<<gGemm, 128, 0, stream>>>(hf, 128, 128, WqT,  qm, nullptr, nullptr, 0, 0);
  gemm_n128<<<gGemm, 128, 0, stream>>>(hf, 128, 128, WkT,  km, nullptr, nullptr, 0, 0);
  gemm_n128<<<gGemm, 128, 0, stream>>>(hf, 128, 128, WvmT, vT, nullptr, nullptr, 0, 1);
  // self-attention
  flash_attn<<<NB * NH * 32, 32, 0, stream>>>(qm, km, vT, Om);
  // h_f <- O @ mma_Wo  (hf buffer reused; q/k now dead)
  gemm_n128<<<gGemm, 128, 0, stream>>>(Om, 128, 128, WoT, hf, nullptr, nullptr, 0, 0);
  // ck = h_v @ mha_Wk ; cv = h_f @ mha_Wv
  gemm_n128<<<gGemm, 128, 0, stream>>>(hv, 128, 128, WkcT, ckm, nullptr, nullptr, 0, 0);
  gemm_n128<<<gGemm, 128, 0, stream>>>(hf, 128, 128, WvcT, cvm, nullptr, nullptr, 0, 0);
  // collapsed cross-attention + tail
  cross_ctx<<<NB, 256, 0, stream>>>(cq, ckm, cvm, ctx);
  tail_kernel<<<NB, 128, 0, stream>>>(ctx, hvmean, mha_Wo, W_ge, b_ge, W_gv, b_gv,
                                      ln_e_g, ln_e_b, ln_v_g, ln_v_b,
                                      W_c1, b_c1, W_c2, b_c2, out);
}
